// EncoderBlock_79422535238225
// MI455X (gfx1250) — matrix operand order repaired and checked by reading, not compiled, not run on hardware
//
#include <hip/hip_runtime.h>

// EncoderBlock for MI455X (gfx1250): bf16 WMMA GEMMs + flash attention
// with TDM (tensor_load_to_lds) staging of K tiles.
// B=1, S=4096, D=768, H=12, HD=64.

#define S_LEN   4096
#define D_MODEL 768
#define N_HEADS 12
#define HEAD_DIM 64

#if defined(__has_builtin)
#if __has_builtin(__builtin_amdgcn_tensor_load_to_lds) && \
    __has_builtin(__builtin_amdgcn_s_wait_tensorcnt)
#define USE_TDM 1
#endif
#endif
#ifndef USE_TDM
#define USE_TDM 0
#endif

typedef __attribute__((ext_vector_type(16))) __bf16 bf16x16;
typedef __attribute__((ext_vector_type(8)))  __bf16 bf16x8;
typedef __attribute__((ext_vector_type(8)))  float  f32x8;

__device__ __forceinline__ __bf16 f2bf(float f) {
  union { float f; unsigned u; } a; a.f = f;
  unsigned r = a.u + 0x7fffu + ((a.u >> 16) & 1u);  // round-to-nearest-even
  unsigned short h = (unsigned short)(r >> 16);
  return __builtin_bit_cast(__bf16, h);
}
__device__ __forceinline__ float bf2f(__bf16 b) {
  unsigned short h = __builtin_bit_cast(unsigned short, b);
  union { unsigned u; float f; } a; a.u = ((unsigned)h) << 16;
  return a.f;
}

// ---- WMMA fragment helpers (layouts per CDNA5 ISA 7.12.2) ----
// A (16x32 bf16, MxK): lane l -> M=l&15; K = (l<16 ? {0..7,16..23} : {8..15,24..31})
__device__ __forceinline__ bf16x16 frag_a_load(const __bf16* base, int stride) {
  const int lane = threadIdx.x & 31;
  const int m  = lane & 15;
  const int ko = (lane >> 4) << 3;           // 0 or 8
  const __bf16* p = base + m * stride + ko;
  bf16x8 lo = *(const bf16x8*)(p);           // K = ko..ko+7
  bf16x8 hi = *(const bf16x8*)(p + 16);      // K = 16+ko..16+ko+7
  bf16x16 f;
#pragma unroll
  for (int i = 0; i < 8; ++i) { f[i] = lo[i]; f[i + 8] = hi[i]; }
  return f;
}
// B (32x16 bf16, KxN) read from an N-major tile Bt[n][k]:
// lane l -> N=l&15; K = (l<16 ? {0..7,16..23} : {8..15,24..31})
__device__ __forceinline__ bf16x16 frag_b_load(const __bf16* base, int stride) {
  const int lane = threadIdx.x & 31;
  const int n  = lane & 15;
  const int kb = (lane >> 4) << 3;           // 0 or 8
  const __bf16* p = base + n * stride + kb;
  bf16x8 lo = *(const bf16x8*)(p);
  bf16x8 hi = *(const bf16x8*)(p + 16);
  bf16x16 f;
#pragma unroll
  for (int i = 0; i < 8; ++i) { f[i] = lo[i]; f[i + 8] = hi[i]; }
  return f;
}
__device__ __forceinline__ f32x8 wmma_bf16(bf16x16 a, bf16x16 b, f32x8 c) {
  return __builtin_amdgcn_wmma_f32_16x16x32_bf16(false, a, false, b,
                                                 (short)0, c, false, false);
}

#if USE_TDM
typedef unsigned int u32x4_t __attribute__((ext_vector_type(4)));
typedef int          i32x8_t __attribute__((ext_vector_type(8)));
typedef int          i32x4_t __attribute__((ext_vector_type(4)));

// TDM 2D tile load: 64x64 bf16 tile, row-major, contiguous (stride==width).
// D# packing per CDNA5 ISA.
// This toolchain's builtin arity is 6: (g0, g1, g2, g3, g4, cpol).
__device__ __forceinline__ void tdm_load_k64(const __bf16* gptr, unsigned lds_off) {
  unsigned long long ga = (unsigned long long)gptr;
  u32x4_t g0;
  g0[0] = 1u;                                            // count=1, user mode
  g0[1] = lds_off;                                       // lds_addr (bytes)
  g0[2] = (unsigned)(ga & 0xffffffffu);                  // global_addr[31:0]
  g0[3] = (unsigned)((ga >> 32) & 0x01ffffffu) | (2u << 30);  // addr[56:32] | type=2
  i32x8_t g1;
  g1[0] = (int)(1u << 16);          // workgroup_mask=0, data_size=1 (2 bytes)
  g1[1] = (int)(64u << 16);         // tensor_dim0[15:0]=64 (bits 63:48)
  g1[2] = (int)(64u << 16);         // tensor_dim0 hi=0 | tensor_dim1[15:0]=64
  g1[3] = (int)(64u << 16);         // tensor_dim1 hi=0 | tile_dim0=64
  g1[4] = (int)64;                  // tile_dim1=64, tile_dim2=0
  g1[5] = (int)64;                  // tensor_dim0_stride=64
  g1[6] = 0;                        // stride hi | tensor_dim1_stride lo (2D: unused)
  g1[7] = 0;
  i32x4_t gz4;
  gz4[0] = 0; gz4[1] = 0; gz4[2] = 0; gz4[3] = 0;        // groups 2/3 unused (2D)
  i32x8_t gz8;
#pragma unroll
  for (int i = 0; i < 8; ++i) gz8[i] = 0;
  __builtin_amdgcn_tensor_load_to_lds(g0, g1, gz4, gz4, gz8, 0);
}
#endif

// ---------------- weight fp32 -> bf16 convert ----------------
__global__ void __launch_bounds__(256) cvt_kernel(const float* __restrict__ in,
                                                  __bf16* __restrict__ out, int n) {
  int i = blockIdx.x * 256 + threadIdx.x;
  if (i < n) out[i] = f2bf(in[i]);
}

// ---------------- LayerNorm (+ optional residual add) ----------------
__global__ void __launch_bounds__(256) ln_kernel(
    const float* __restrict__ x, const float* __restrict__ add,
    float* __restrict__ x1_out, __bf16* __restrict__ hout,
    const float* __restrict__ g, const float* __restrict__ b) {
  const int row = blockIdx.x;
  const size_t base = (size_t)row * D_MODEL;
  __shared__ float red[16];
  float v[3];
  float s = 0.f, s2 = 0.f;
#pragma unroll
  for (int i = 0; i < 3; ++i) {
    int c = threadIdx.x + i * 256;
    float t = x[base + c];
    if (add) t += add[base + c];
    v[i] = t; s += t; s2 += t * t;
  }
#pragma unroll
  for (int m = 1; m < 32; m <<= 1) { s += __shfl_xor(s, m, 32); s2 += __shfl_xor(s2, m, 32); }
  const int wid = threadIdx.x >> 5, lane = threadIdx.x & 31;
  if (lane == 0) { red[wid] = s; red[8 + wid] = s2; }
  __syncthreads();
  if (wid == 0) {
    float a  = (lane < 8) ? red[lane] : 0.f;
    float a2 = (lane < 8) ? red[8 + lane] : 0.f;
#pragma unroll
    for (int m = 1; m < 8; m <<= 1) { a += __shfl_xor(a, m, 32); a2 += __shfl_xor(a2, m, 32); }
    if (lane == 0) { red[0] = a; red[1] = a2; }
  }
  __syncthreads();
  const float mu  = red[0] * (1.f / D_MODEL);
  const float var = red[1] * (1.f / D_MODEL) - mu * mu;
  const float rstd = rsqrtf(var + 1e-5f);
#pragma unroll
  for (int i = 0; i < 3; ++i) {
    int c = threadIdx.x + i * 256;
    if (x1_out) x1_out[base + c] = v[i];
    hout[base + c] = f2bf((v[i] - mu) * rstd * g[c] + b[c]);
  }
}

// ---------------- tiled bf16 GEMM: C[M,N] = A[M,768] * Bw[768,N] ----------------
// EPI 0: +bias, scatter bf16 into q/k/v head-major [H][S][64]
// EPI 1: +bias +resid, write fp32 outp[M,768]
template <int EPI>
__global__ void __launch_bounds__(256) gemm_kernel(
    const __bf16* __restrict__ A, const __bf16* __restrict__ Bw,
    const float* __restrict__ bias, int N,
    __bf16* __restrict__ qb, __bf16* __restrict__ kb, __bf16* __restrict__ vb,
    const float* __restrict__ resid, float* __restrict__ outp) {
  __shared__ __bf16 As[128][40];   // row-major M x K tile (pad 32->40)
  __shared__ __bf16 Bs[128][40];   // transposed: [n][k]
  const int bm = blockIdx.y * 128;
  const int bn = blockIdx.x * 128;
  const int tid = threadIdx.x;
  const int lane = tid & 31;
  const int wid = tid >> 5;
  const int wm = (wid & 3) * 32;   // 4 waves along M
  const int wn = (wid >> 2) * 64;  // 2 waves along N

  f32x8 acc[2][4];
#pragma unroll
  for (int t = 0; t < 2; ++t)
#pragma unroll
    for (int g = 0; g < 4; ++g)
#pragma unroll
      for (int r = 0; r < 8; ++r) acc[t][g][r] = 0.0f;

  for (int k0 = 0; k0 < 768; k0 += 32) {
    if (k0 + 32 < 768) {  // prefetch next tiles -> global_prefetch_b8
      __builtin_prefetch(&A[(size_t)(bm + (tid >> 1)) * 768 + k0 + 32], 0, 3);
      __builtin_prefetch(&Bw[(size_t)(k0 + 32 + (tid >> 3)) * N + bn], 0, 3);
    }
#pragma unroll
    for (int i = 0; i < 2; ++i) {  // A tile: 128x32, 8 contiguous per slot
      int idx = tid + i * 256;
      int r = idx >> 2;
      int c = (idx & 3) << 3;
      *(bf16x8*)&As[r][c] = *(const bf16x8*)&A[(size_t)(bm + r) * 768 + k0 + c];
    }
#pragma unroll
    for (int i = 0; i < 2; ++i) {  // B tile: 32x128, store transposed
      int idx = tid + i * 256;
      int kk = idx >> 4;
      int nn = (idx & 15) << 3;
      bf16x8 vv = *(const bf16x8*)&Bw[(size_t)(k0 + kk) * N + bn + nn];
#pragma unroll
      for (int j = 0; j < 8; ++j) Bs[nn + j][kk] = vv[j];
    }
    __syncthreads();
    bf16x16 af[2];
#pragma unroll
    for (int t = 0; t < 2; ++t) af[t] = frag_a_load(&As[wm + t * 16][0], 40);
#pragma unroll
    for (int g = 0; g < 4; ++g) {
      bf16x16 bfr = frag_b_load(&Bs[wn + g * 16][0], 40);
#pragma unroll
      for (int t = 0; t < 2; ++t) acc[t][g] = wmma_bf16(af[t], bfr, acc[t][g]);
    }
    __syncthreads();
  }

#pragma unroll
  for (int t = 0; t < 2; ++t)
#pragma unroll
    for (int g = 0; g < 4; ++g)
#pragma unroll
      for (int r = 0; r < 8; ++r) {
        int row = bm + wm + t * 16 + r + ((lane >> 4) << 3);
        int col = bn + wn + g * 16 + (lane & 15);
        float v = acc[t][g][r] + bias[col];
        if (EPI == 0) {
          int part = col / D_MODEL;           // 0=q 1=k 2=v
          int cc = col - part * D_MODEL;
          int head = cc >> 6;
          int hd = cc & 63;
          __bf16* dst = (part == 0) ? qb : ((part == 1) ? kb : vb);
          dst[((size_t)(head * S_LEN + row) << 6) + hd] = f2bf(v);
        } else {
          size_t o = (size_t)row * D_MODEL + col;
          outp[o] = v + resid[o];
        }
      }
}

// ---------------- flash attention: one head x 128 q-rows per block ----------------
// 64-key tiles: halves softmax/rescale cost per key vs 32-key tiles.
__global__ void __launch_bounds__(256) flash_kernel(
    const __bf16* __restrict__ qbuf, const __bf16* __restrict__ kbuf,
    const __bf16* __restrict__ vbuf, float* __restrict__ obuf /* [S][768] */) {
  __shared__ __bf16 Kt[64][64];       // [key][hd]          (8 KB)
  __shared__ __bf16 Vt[64][64];       // [hd][key] transposed (8 KB)
  __shared__ __bf16 Pb[8][16][64];    // per-wave P staging  (16 KB)
  const int head = blockIdx.y;
  const int q0 = blockIdx.x * 128;
  const int tid = threadIdx.x, wid = tid >> 5, lane = tid & 31;
  const int qbase = q0 + wid * 16;

  // Q fragments (K-dim = HD = 64 -> 2 chunks), pre-scaled by 1/sqrt(64)
  const __bf16* qp = qbuf + ((size_t)(head * S_LEN + qbase) << 6);
  bf16x16 qa[2];
#pragma unroll
  for (int c = 0; c < 2; ++c) {
    bf16x16 f = frag_a_load(qp + c * 32, 64);
#pragma unroll
    for (int i = 0; i < 16; ++i) f[i] = f2bf(bf2f(f[i]) * 0.125f);
    qa[c] = f;
  }

  f32x8 O[4];
#pragma unroll
  for (int g = 0; g < 4; ++g)
#pragma unroll
    for (int r = 0; r < 8; ++r) O[g][r] = 0.0f;
  float mrow[8], lrow[8];
#pragma unroll
  for (int r = 0; r < 8; ++r) { mrow[r] = -1e30f; lrow[r] = 0.0f; }

  for (int kb0 = 0; kb0 < S_LEN; kb0 += 64) {
    const __bf16* kg = kbuf + ((size_t)(head * S_LEN + kb0) << 6);
    const __bf16* vg = vbuf + ((size_t)(head * S_LEN + kb0) << 6);
#if USE_TDM
    if (wid == 0) {                      // one wave issues the DMA (EXEC ignored)
      tdm_load_k64(kg, (unsigned)(size_t)&Kt[0][0]);
      __builtin_amdgcn_s_wait_tensorcnt(0);
    }
#endif
#pragma unroll
    for (int i = 0; i < 2; ++i) {        // V: load + transpose into Vt[hd][key]
      int idx = tid + i * 256;           // 0..511 -> 64 rows x 8 chunks
      int r = idx >> 3;
      int c = (idx & 7) << 3;
      bf16x8 vv = *(const bf16x8*)&vg[((size_t)r << 6) + c];
#pragma unroll
      for (int j = 0; j < 8; ++j) Vt[c + j][r] = vv[j];
#if !USE_TDM
      *(bf16x8*)&Kt[r][c] = *(const bf16x8*)&kg[((size_t)r << 6) + c];
#endif
    }
    __syncthreads();

    // scores: S[16q x 64k] = Q[16x64] * K^T, 4 key-groups x 2 hd-chunks
    f32x8 sc[4];
#pragma unroll
    for (int g = 0; g < 4; ++g) {
#pragma unroll
      for (int r = 0; r < 8; ++r) sc[g][r] = 0.0f;
#pragma unroll
      for (int c = 0; c < 2; ++c) {
        bf16x16 bfr = frag_b_load(&Kt[g * 16][c * 32], 64);
        sc[g] = wmma_bf16(qa[c], bfr, sc[g]);
      }
    }

    // online softmax (rows in 16-lane halves; reduce via shfl_xor)
#pragma unroll
    for (int r = 0; r < 8; ++r) {
      float s0 = sc[0][r], s1 = sc[1][r], s2 = sc[2][r], s3 = sc[3][r];
      float mx = fmaxf(fmaxf(s0, s1), fmaxf(s2, s3));
#pragma unroll
      for (int m = 1; m < 16; m <<= 1) mx = fmaxf(mx, __shfl_xor(mx, m, 32));
      float mn = fmaxf(mrow[r], mx);
      float p0 = __expf(s0 - mn), p1 = __expf(s1 - mn);
      float p2 = __expf(s2 - mn), p3 = __expf(s3 - mn);
      float rs = (p0 + p1) + (p2 + p3);
#pragma unroll
      for (int m = 1; m < 16; m <<= 1) rs += __shfl_xor(rs, m, 32);
      float alpha = __expf(mrow[r] - mn);
      lrow[r] = lrow[r] * alpha + rs;
      mrow[r] = mn;
#pragma unroll
      for (int g = 0; g < 4; ++g) O[g][r] *= alpha;
      int M = r + ((lane >> 4) << 3);
      int n = lane & 15;
      Pb[wid][M][n]      = f2bf(p0);
      Pb[wid][M][16 + n] = f2bf(p1);
      Pb[wid][M][32 + n] = f2bf(p2);
      Pb[wid][M][48 + n] = f2bf(p3);
    }

    // P as two A-fragments (keys 0..31, 32..63); wave-private LDS roundtrip
    bf16x16 pa0 = frag_a_load(&Pb[wid][0][0], 64);
    bf16x16 pa1 = frag_a_load(&Pb[wid][0][32], 64);
#pragma unroll
    for (int g = 0; g < 4; ++g) {
      O[g] = wmma_bf16(pa0, frag_b_load(&Vt[g * 16][0], 64), O[g]);
      O[g] = wmma_bf16(pa1, frag_b_load(&Vt[g * 16][32], 64), O[g]);
    }
    __syncthreads();
  }

#pragma unroll
  for (int g = 0; g < 4; ++g)
#pragma unroll
    for (int r = 0; r < 8; ++r) {
      int row = qbase + r + ((lane >> 4) << 3);
      int col = head * HEAD_DIM + g * 16 + (lane & 15);
      obuf[(size_t)row * D_MODEL + col] = O[g][r] / lrow[r];
    }
}

// ---------------- host side ----------------
extern "C" void kernel_launch(void* const* d_in, const int* in_sizes, int n_in,
                              void* d_out, int out_size, void* d_ws, size_t ws_size,
                              hipStream_t stream) {
  (void)in_sizes; (void)n_in; (void)out_size; (void)ws_size;
  const float* x     = (const float*)d_in[0];
  const float* w_qkv = (const float*)d_in[1];
  const float* b_qkv = (const float*)d_in[2];
  const float* w_mlp = (const float*)d_in[3];
  const float* b_mlp = (const float*)d_in[4];
  const float* ln_g  = (const float*)d_in[5];
  const float* ln_b  = (const float*)d_in[6];
  float* out = (float*)d_out;

  const size_t SD = (size_t)S_LEN * D_MODEL;
  char* ws = (char*)d_ws;
  size_t off = 0;
  auto alloc = [&](size_t bytes) -> char* {
    char* p = ws + off;
    off += (bytes + 255) & ~(size_t)255;
    return p;
  };
  __bf16* h1    = (__bf16*)alloc(SD * 2);
  __bf16* wqkvb = (__bf16*)alloc((size_t)D_MODEL * 3 * D_MODEL * 2);
  __bf16* wmlpb = (__bf16*)alloc((size_t)D_MODEL * D_MODEL * 2);
  __bf16* qb    = (__bf16*)alloc(SD * 2);
  __bf16* kb    = (__bf16*)alloc(SD * 2);
  __bf16* vb    = (__bf16*)alloc(SD * 2);
  float*  obuf  = (float*)alloc(SD * 4);
  float*  x1    = (float*)alloc(SD * 4);
  __bf16* h2    = (__bf16*)alloc(SD * 2);

  const int nqkv = D_MODEL * 3 * D_MODEL;
  const int nmlp = D_MODEL * D_MODEL;
  cvt_kernel<<<(nqkv + 255) / 256, 256, 0, stream>>>(w_qkv, wqkvb, nqkv);
  cvt_kernel<<<(nmlp + 255) / 256, 256, 0, stream>>>(w_mlp, wmlpb, nmlp);

  // h1 = LN(x)
  ln_kernel<<<S_LEN, 256, 0, stream>>>(x, nullptr, nullptr, h1, ln_g, ln_b);
  // qkv = h1 @ w_qkv + b_qkv, scattered to head-major bf16 q/k/v
  gemm_kernel<0><<<dim3(3 * D_MODEL / 128, S_LEN / 128), 256, 0, stream>>>(
      h1, wqkvb, b_qkv, 3 * D_MODEL, qb, kb, vb, nullptr, nullptr);
  // o = softmax(qk^T/8) v  (flash; never materializes [H,S,S])
  flash_kernel<<<dim3(S_LEN / 128, N_HEADS), 256, 0, stream>>>(qb, kb, vb, obuf);
  // x1 = o + x ; h2 = LN(x1)
  ln_kernel<<<S_LEN, 256, 0, stream>>>(obuf, x, x1, h2, ln_g, ln_b);
  // out = h2 @ w_mlp + b_mlp + x1
  gemm_kernel<1><<<dim3(D_MODEL / 128, S_LEN / 128), 256, 0, stream>>>(
      h2, wmlpb, b_mlp, D_MODEL, nullptr, nullptr, nullptr, x1, out);
}
